// MultiHeadAttention_44066364457464
// MI455X (gfx1250) — matrix operand order repaired and checked by reading, not compiled, not run on hardware
//
#include <hip/hip_runtime.h>
#include <hip/hip_bf16.h>

// Problem constants (fixed by the reference)
constexpr int kB = 2;
constexpr int kL = 2048;
constexpr int kC = 1024;
constexpr int kH = 16;
constexpr int kD = 64;                 // kC / kH
constexpr int kM = kB * kL;            // 4096 GEMM rows

typedef __attribute__((ext_vector_type(16))) _Float16 v16h;
typedef __attribute__((ext_vector_type(8)))  _Float16 v8h;
typedef __attribute__((ext_vector_type(8)))  float    v8f;

__device__ __forceinline__ v16h mk_frag(const _Float16* p0, const _Float16* p1) {
    v8h lo = *(const v8h*)p0;
    v8h hi = *(const v8h*)p1;
    v16h r;
#pragma unroll
    for (int i = 0; i < 8; ++i) { r[i] = lo[i]; r[i + 8] = hi[i]; }
    return r;
}

__device__ __forceinline__ v8f wmma_f16(v16h a, v16h b, v8f c) {
    return __builtin_amdgcn_wmma_f32_16x16x32_f16(
        /*neg_a=*/false, a, /*neg_b=*/false, b,
        /*c_mod=*/(short)0, c, /*reuse_a=*/false, /*reuse_b=*/false);
}

// Async global -> LDS staging (gfx1250, tracked with ASYNCcnt)
__device__ __forceinline__ void async_ld_b128(const void* gaddr, void* ldsaddr) {
    unsigned lo = (unsigned)(size_t)ldsaddr;                 // LDS byte offset
    unsigned long long ga = (unsigned long long)(size_t)gaddr;
    asm volatile("global_load_async_to_lds_b128 %0, %1, off"
                 :: "v"(lo), "v"(ga) : "memory");
}

// ---------------------------------------------------------------------------
// f32 -> f16 elementwise convert
// ---------------------------------------------------------------------------
__global__ __launch_bounds__(256) void cvt_f16_kernel(const float* __restrict__ x,
                                                      _Float16* __restrict__ y, int n) {
    int i = blockIdx.x * 256 + threadIdx.x;
    if (i < n) y[i] = (_Float16)x[i];
}

// ---------------------------------------------------------------------------
// Weight transpose + convert: Wt[n][k] = (f16) W[k][n], both kC x kC
// ---------------------------------------------------------------------------
__global__ __launch_bounds__(256) void wtrans_kernel(const float* __restrict__ W,
                                                     _Float16* __restrict__ Wt) {
    int i = blockIdx.x * 256 + threadIdx.x;   // i = n*kC + k
    int n = i >> 10;
    int k = i & (kC - 1);
    Wt[i] = (_Float16)W[(size_t)k * kC + n];
}

// ---------------------------------------------------------------------------
// GEMM: out[M x N] = A[M x K] * Bt[N x K]^T   (Bt is K-major per output col)
// One wave -> 32x64 macro-tile (2x4 grid of 16x16 WMMA tiles) for fragment
// reuse: 12 b128 loads feed 8 WMMAs per k-step.
// mode 0: f16 row-major; mode 1: f16 transposed per-batch (Vt[b][n][l]);
// mode 2: f32 row-major (final projection).
// ---------------------------------------------------------------------------
__global__ __launch_bounds__(256) void gemm16_kernel(const _Float16* __restrict__ A,
                                                     const _Float16* __restrict__ Bt,
                                                     void* __restrict__ outp,
                                                     int M, int N, int K,
                                                     int mode, int Lsub) {
    const int lane = threadIdx.x & 31;
    const int w    = threadIdx.x >> 5;
    const int hi   = lane >> 4;
    const int ln   = lane & 15;

    const int tilesN = N >> 6;                 // 64-wide macro columns
    const int macro  = blockIdx.x * 8 + w;
    const int m0     = (macro / tilesN) * 32;
    const int n0     = (macro % tilesN) * 64;

    const _Float16* Ar0 = A + (size_t)(m0 + ln) * K;
    const _Float16* Ar1 = A + (size_t)(m0 + 16 + ln) * K;
    const _Float16* Br0 = Bt + (size_t)(n0 + ln) * K;
    const _Float16* Br1 = Bt + (size_t)(n0 + 16 + ln) * K;
    const _Float16* Br2 = Bt + (size_t)(n0 + 32 + ln) * K;
    const _Float16* Br3 = Bt + (size_t)(n0 + 48 + ln) * K;

    v8f acc[2][4] = {};
#pragma unroll 2
    for (int kk = 0; kk < K; kk += 32) {
        v16h a0 = mk_frag(Ar0 + kk + 16 * hi, Ar0 + kk + 16 * hi + 8);
        v16h a1 = mk_frag(Ar1 + kk + 16 * hi, Ar1 + kk + 16 * hi + 8);
        v16h b0 = mk_frag(Br0 + kk + 16 * hi, Br0 + kk + 16 * hi + 8);
        v16h b1 = mk_frag(Br1 + kk + 16 * hi, Br1 + kk + 16 * hi + 8);
        v16h b2 = mk_frag(Br2 + kk + 16 * hi, Br2 + kk + 16 * hi + 8);
        v16h b3 = mk_frag(Br3 + kk + 16 * hi, Br3 + kk + 16 * hi + 8);
        acc[0][0] = wmma_f16(a0, b0, acc[0][0]);
        acc[1][0] = wmma_f16(a1, b0, acc[1][0]);
        acc[0][1] = wmma_f16(a0, b1, acc[0][1]);
        acc[1][1] = wmma_f16(a1, b1, acc[1][1]);
        acc[0][2] = wmma_f16(a0, b2, acc[0][2]);
        acc[1][2] = wmma_f16(a1, b2, acc[1][2]);
        acc[0][3] = wmma_f16(a0, b3, acc[0][3]);
        acc[1][3] = wmma_f16(a1, b3, acc[1][3]);
    }

#pragma unroll
    for (int i = 0; i < 2; ++i)
#pragma unroll
        for (int j = 0; j < 4; ++j) {
            const int n = n0 + j * 16 + ln;
            if (mode == 0) {
                _Float16* o = (_Float16*)outp;
#pragma unroll
                for (int r = 0; r < 8; ++r) {
                    int m = m0 + i * 16 + r + 8 * hi;
                    o[(size_t)m * N + n] = (_Float16)acc[i][j][r];
                }
            } else if (mode == 1) {
                _Float16* o = (_Float16*)outp;
#pragma unroll
                for (int r = 0; r < 8; ++r) {
                    int m  = m0 + i * 16 + r + 8 * hi;
                    int bb = m / Lsub;
                    int l  = m % Lsub;
                    o[((size_t)bb * N + n) * Lsub + l] = (_Float16)acc[i][j][r];
                }
            } else {
                float* o = (float*)outp;
#pragma unroll
                for (int r = 0; r < 8; ++r) {
                    int m = m0 + i * 16 + r + 8 * hi;
                    o[(size_t)m * N + n] = acc[i][j][r];
                }
            }
        }
}

// ---------------------------------------------------------------------------
// Fused flash attention. Grid (kL/64, kH, kB), block 128 (4 waves).
// Each wave: 16 query rows, online softmax over 32-key blocks.
// K/V tiles staged block-wide in LDS via global_load_async_to_lds_b128.
// Q,K: f16 [B,L,C]; Vt: f16 [B,C,L]; att: f16 [B,L,C].
// ---------------------------------------------------------------------------
__global__ __launch_bounds__(128) void attn_kernel(const _Float16* __restrict__ Q,
                                                   const _Float16* __restrict__ K,
                                                   const _Float16* __restrict__ Vt,
                                                   const unsigned char* __restrict__ mask,
                                                   _Float16* __restrict__ att) {
    __shared__ alignas(16) _Float16 Klds[32][80];    // 32 keys x 64 d (+16 pad)
    __shared__ alignas(16) _Float16 Vlds[64][40];    // 64 d   x 32 keys (+8 pad)
    __shared__ alignas(16) _Float16 Pbuf[4][16][40]; // per-wave 16x32 (+8 pad)

    const int tid  = threadIdx.x;
    const int lane = tid & 31;
    const int w    = tid >> 5;
    const int hi   = lane >> 4;
    const int ln   = lane & 15;

    const int qb = blockIdx.x * 64 + w * 16;
    const int h  = blockIdx.y;
    const int b  = blockIdx.z;

    // Q A-fragments for the two 32-wide d-chunks, pre-scaled by 1/sqrt(D)=0.125
    const _Float16* Qrow = Q + ((size_t)(b * kL + qb + ln)) * kC + h * kD;
    v16h aq0 = mk_frag(Qrow + 16 * hi,      Qrow + 16 * hi + 8);
    v16h aq1 = mk_frag(Qrow + 32 + 16 * hi, Qrow + 32 + 16 * hi + 8);
    {
        const _Float16 sc = (_Float16)0.125f;
#pragma unroll
        for (int i = 0; i < 16; ++i) { aq0[i] *= sc; aq1[i] *= sc; }
    }

    float m_i[8], l_i[8];
#pragma unroll
    for (int r = 0; r < 8; ++r) { m_i[r] = -3.0e38f; l_i[r] = 0.0f; }
    v8f oacc[4] = {};

    const unsigned char* mrow = mask + (size_t)b * kL * kL;

    for (int kb = 0; kb < kL; kb += 32) {
        __syncthreads();   // previous iteration's LDS reads done before overwrite

        // ---- async stage K tile (32x64 f16) and V tile (64x32 f16) ----
#pragma unroll
        for (int rep = 0; rep < 2; ++rep) {
            int c    = tid + rep * 128;         // 256 16B chunks
            int key  = c >> 3;
            int doff = (c & 7) * 8;
            const _Float16* g = K + ((size_t)(b * kL + kb + key)) * kC + h * kD + doff;
            async_ld_b128(g, &Klds[key][doff]);
        }
#pragma unroll
        for (int rep = 0; rep < 2; ++rep) {
            int c    = tid + rep * 128;
            int d    = c >> 2;
            int koff = (c & 3) * 8;
            const _Float16* g = Vt + ((size_t)(b * kC + h * kD + d)) * kL + kb + koff;
            async_ld_b128(g, &Vlds[d][koff]);
        }
        asm volatile("s_wait_asynccnt 0x0" ::: "memory");
        __syncthreads();   // all waves' async data visible

        // ---- S = (Q/sqrt(D)) * K^T, two 16x16 tiles covering 32 keys ----
        v8f s[2];
#pragma unroll
        for (int t = 0; t < 2; ++t) {
            const int keyl = t * 16 + ln;
            v16h bk0 = mk_frag(&Klds[keyl][16 * hi],      &Klds[keyl][16 * hi + 8]);
            v16h bk1 = mk_frag(&Klds[keyl][32 + 16 * hi], &Klds[keyl][32 + 16 * hi + 8]);
            v8f c = {};
            c = wmma_f16(aq0, bk0, c);
            c = wmma_f16(aq1, bk1, c);
            // additive mask bias (-1e9 where mask false)
            const int key = kb + keyl;
#pragma unroll
            for (int r = 0; r < 8; ++r) {
                int q = qb + r + 8 * hi;
                if (!mrow[(size_t)q * kL + key]) c[r] += -1.0e9f;
            }
            s[t] = c;
        }

        // ---- online softmax row stats (rows live in 16-lane halves) ----
        float rm[8];
#pragma unroll
        for (int r = 0; r < 8; ++r) rm[r] = fmaxf(s[0][r], s[1][r]);
#pragma unroll
        for (int msk = 1; msk < 16; msk <<= 1)
#pragma unroll
            for (int r = 0; r < 8; ++r) rm[r] = fmaxf(rm[r], __shfl_xor(rm[r], msk, 16));

        float corr[8];
#pragma unroll
        for (int r = 0; r < 8; ++r) {
            float mn = fmaxf(m_i[r], rm[r]);
            corr[r]  = __expf(m_i[r] - mn);
            m_i[r]   = mn;
        }
#pragma unroll
        for (int t = 0; t < 2; ++t)
#pragma unroll
            for (int r = 0; r < 8; ++r) s[t][r] = __expf(s[t][r] - m_i[r]);

        float rs[8];
#pragma unroll
        for (int r = 0; r < 8; ++r) rs[r] = s[0][r] + s[1][r];
#pragma unroll
        for (int msk = 1; msk < 16; msk <<= 1)
#pragma unroll
            for (int r = 0; r < 8; ++r) rs[r] += __shfl_xor(rs[r], msk, 16);
#pragma unroll
        for (int r = 0; r < 8; ++r) l_i[r] = l_i[r] * corr[r] + rs[r];

        // ---- shuffle P (C-layout f32) -> A-layout f16 via per-wave LDS ----
#pragma unroll
        for (int t = 0; t < 2; ++t)
#pragma unroll
            for (int r = 0; r < 8; ++r)
                Pbuf[w][r + 8 * hi][t * 16 + ln] = (_Float16)s[t][r];
        asm volatile("s_wait_dscnt 0" ::: "memory");
        const _Float16* pp = &Pbuf[w][ln][0];
        v16h pa = mk_frag(pp + 16 * hi, pp + 16 * hi + 8);

        // ---- O = O*corr + P * V (V B-frags from LDS) ----
#pragma unroll
        for (int u = 0; u < 4; ++u) {
            const int d = u * 16 + ln;
            v16h bv = mk_frag(&Vlds[d][16 * hi], &Vlds[d][16 * hi + 8]);
#pragma unroll
            for (int r = 0; r < 8; ++r) oacc[u][r] *= corr[r];
            oacc[u] = wmma_f16(pa, bv, oacc[u]);
        }
    }

    // ---- epilogue: normalize and store combined heads [B,L,C] ----
#pragma unroll
    for (int u = 0; u < 4; ++u)
#pragma unroll
        for (int r = 0; r < 8; ++r) {
            int q = qb + r + 8 * hi;
            int d = u * 16 + ln;
            att[((size_t)(b * kL + q)) * kC + h * kD + d] =
                (_Float16)(oacc[u][r] / l_i[r]);
        }
}

// ---------------------------------------------------------------------------
// Host launcher
// ---------------------------------------------------------------------------
extern "C" void kernel_launch(void* const* d_in, const int* in_sizes, int n_in,
                              void* d_out, int out_size, void* d_ws, size_t ws_size,
                              hipStream_t stream) {
    (void)in_sizes; (void)n_in; (void)out_size; (void)ws_size;

    const float* Xq = (const float*)d_in[0];
    const float* Xm = (const float*)d_in[1];
    const unsigned char* mask = (const unsigned char*)d_in[2];
    const float* Wq = (const float*)d_in[3];
    const float* Wk = (const float*)d_in[4];
    const float* Wv = (const float*)d_in[5];
    const float* Wo = (const float*)d_in[6];

    _Float16* ws = (_Float16*)d_ws;
    const size_t NX = (size_t)kM * kC;   // 4,194,304
    const size_t NW = (size_t)kC * kC;   // 1,048,576
    _Float16* Xq16  = ws;
    _Float16* Xm16  = Xq16 + NX;
    _Float16* Wqt   = Xm16 + NX;
    _Float16* Wkt   = Wqt + NW;
    _Float16* Wvt   = Wkt + NW;
    _Float16* Wot   = Wvt + NW;
    _Float16* Q16   = Wot + NW;
    _Float16* K16   = Q16 + NX;
    _Float16* Vt16  = K16 + NX;
    _Float16* att16 = Vt16 + NX;

    // 1) convert activations to f16
    cvt_f16_kernel<<<(int)(NX / 256), 256, 0, stream>>>(Xq, Xq16, (int)NX);
    cvt_f16_kernel<<<(int)(NX / 256), 256, 0, stream>>>(Xm, Xm16, (int)NX);

    // 2) transpose+convert weights (K-major for WMMA B operands)
    wtrans_kernel<<<(int)(NW / 256), 256, 0, stream>>>(Wq, Wqt);
    wtrans_kernel<<<(int)(NW / 256), 256, 0, stream>>>(Wk, Wkt);
    wtrans_kernel<<<(int)(NW / 256), 256, 0, stream>>>(Wv, Wvt);
    wtrans_kernel<<<(int)(NW / 256), 256, 0, stream>>>(Wo, Wot);

    // 3) projections: Q, K row-major; V stored transposed per batch [b][c][l]
    const int gemmGrid = (kM / 32) * (kC / 64) / 8;   // 256
    gemm16_kernel<<<gemmGrid, 256, 0, stream>>>(Xq16, Wqt, Q16,  kM, kC, kC, 0, kL);
    gemm16_kernel<<<gemmGrid, 256, 0, stream>>>(Xm16, Wkt, K16,  kM, kC, kC, 0, kL);
    gemm16_kernel<<<gemmGrid, 256, 0, stream>>>(Xm16, Wvt, Vt16, kM, kC, kC, 1, kL);

    // 4) fused flash attention -> att16 [B,L,C]
    dim3 ag(kL / 64, kH, kB);
    attn_kernel<<<ag, 128, 0, stream>>>(Q16, K16, Vt16, mask, att16);

    // 5) output projection -> f32 d_out [B,L,C]
    gemm16_kernel<<<gemmGrid, 256, 0, stream>>>(att16, Wot, d_out, kM, kC, kC, 2, kL);
}
